// Encoder_3556232922499
// MI455X (gfx1250) — matrix operand order repaired and checked by reading, not compiled, not run on hardware
//
#include <hip/hip_runtime.h>
#include <hip/hip_bf16.h>

#define B_  256
#define TM1 255
#define N_  256
#define H_  256

typedef __attribute__((ext_vector_type(16))) __bf16 v16bf;
typedef __attribute__((ext_vector_type(8)))  __bf16 v8bf;
typedef __attribute__((ext_vector_type(8)))  float  v8f;

// Defeat loop-invariant hoisting of the weight loads (round-1 showed the
// compiler caching all 128 B-fragments -> scratch spills). Making the K
// offset opaque keeps the loads inside the loop, while the *pointer* keeps
// its kernel-arg provenance so loads lower to global_load (not flat_load,
// which would tie up DScnt/LDS path alongside our ds_load A-fragments).
#define NOHOIST_I32(x) asm volatile("" : "+s"(x))

__device__ __forceinline__ unsigned short f2bf(float f) {
    union { float f; unsigned u; } c; c.f = f;
    unsigned u = c.u;
    u += 0x7fffu + ((u >> 16) & 1u);   // round-to-nearest-even
    return (unsigned short)(u >> 16);
}
__device__ __forceinline__ float sigmoidf(float x) {
    return 1.0f / (1.0f + __expf(-x));
}

// ---------------------------------------------------------------------------
// Kernel 1: x_score = sum_t X[b,t,n]*w_x[t] + bias; alpha = softmax_n(x_score)
// (the hs[b] term in the reference is a per-row constant -> softmax-invariant)
// ---------------------------------------------------------------------------
__global__ void alpha_kernel(const float* __restrict__ X,
                             const float* __restrict__ attn_w,
                             const float* __restrict__ attn_b,
                             float* __restrict__ alpha) {
    __shared__ float wx[TM1];
    __shared__ float red[256];
    const int b = blockIdx.x;
    const int n = threadIdx.x;
    if (n < TM1) wx[n] = attn_w[2 * H_ + n];
    __syncthreads();

    const float* Xb = X + (size_t)b * TM1 * N_ + n;
    float s = attn_b[0];
    for (int t = 0; t < TM1; ++t) s += Xb[(size_t)t * N_] * wx[t];

    red[n] = s; __syncthreads();
    for (int off = 128; off > 0; off >>= 1) {
        if (n < off) red[n] = fmaxf(red[n], red[n + off]);
        __syncthreads();
    }
    float m = red[0]; __syncthreads();
    float e = __expf(s - m);
    red[n] = e; __syncthreads();
    for (int off = 128; off > 0; off >>= 1) {
        if (n < off) red[n] += red[n + off];
        __syncthreads();
    }
    alpha[b * N_ + n] = e / red[0];
}

// ---------------------------------------------------------------------------
// Kernel 2: weights f32 -> bf16 (row-major == WMMA B-fragment order, since
// gates = x~ @ W^T means B[k=n][col=j] = W[j][n])
// ---------------------------------------------------------------------------
__global__ void cvt_kernel(const float* __restrict__ Wih,
                           const float* __restrict__ Whh,
                           unsigned short* __restrict__ WihBf,
                           unsigned short* __restrict__ WhhBf) {
    int i = blockIdx.x * blockDim.x + threadIdx.x;   // 0 .. 4*H*N-1
    WihBf[i] = f2bf(Wih[i]);
    WhhBf[i] = f2bf(Whh[i]);
}

// ---------------------------------------------------------------------------
// Kernel 3: persistent batch-partitioned LSTM recurrence.
// grid = 16 blocks x 256 threads (8 waves). Block owns 16 batch rows for all
// 255 steps. Wave w owns hidden units [32w, 32w+32): its 8 output tiles are
// (gate g, sub-tile h2) -> all four gates of each hidden unit live in one
// wave; the c-state never leaves accumulator-layout registers.
// ---------------------------------------------------------------------------
#define AXS 264   // padded LDS row stride (ushort): 528B -> conflict-free rows

__global__ void __launch_bounds__(256, 1)
lstm_kernel(const float* __restrict__ X,
            const float* __restrict__ alpha,
            const unsigned short* __restrict__ WihBf,
            const unsigned short* __restrict__ WhhBf,
            const float* __restrict__ b_ih,
            const float* __restrict__ b_hh,
            float* __restrict__ Xtilde,
            float* __restrict__ Xenc) {
    __shared__ unsigned short Ax[16 * AXS];   // x~ tile, bf16
    __shared__ unsigned short Ah[16 * AXS];   // h    tile, bf16

    const int tid  = threadIdx.x;
    const int lane = tid & 31;
    const int wv   = tid >> 5;       // wave 0..7
    const int ln16 = lane & 15;
    const int hi   = lane >> 4;      // lane half
    const int b0   = blockIdx.x * 16;

    // alpha for this thread's column (n = tid) across the block's 16 rows
    float areg[16];
    #pragma unroll
    for (int r = 0; r < 16; ++r) areg[r] = alpha[(b0 + r) * N_ + tid];

    // per-lane weight-tile element offsets (B fragment: 16 contiguous bf16 of
    // row `col`, K-half selected by lane half) and bias per output tile
    float biasv[4][2];
    int   colOff[4][2];
    const int khB = hi * 16;
    #pragma unroll
    for (int g = 0; g < 4; ++g)
        #pragma unroll
        for (int h2 = 0; h2 < 2; ++h2) {
            int col = g * H_ + wv * 32 + h2 * 16 + ln16;
            biasv[g][h2]  = b_ih[col] + b_hh[col];
            colOff[g][h2] = col * N_ + khB;        // N_ == H_ == 256
        }

    // h0 = 0, c0 = 0
    for (int i = tid; i < 16 * AXS; i += 256) Ah[i] = 0;
    v8f cfrag[2];
    #pragma unroll
    for (int h2 = 0; h2 < 2; ++h2)
        #pragma unroll
        for (int r = 0; r < 8; ++r) cfrag[h2][r] = 0.0f;

    const int kbA  = hi * 16;   // A-fragment K base within a 32-chunk
    const int rowA = ln16;

    __syncthreads();

    for (int t = 0; t < TM1; ++t) {
        // ---- stage x_tilde = alpha * x_t (thread tid = column n), emit output
        #pragma unroll 4
        for (int r = 0; r < 16; ++r) {
            size_t gi = (size_t)(b0 + r) * TM1 * N_ + (size_t)t * N_ + tid;
            float xv = __builtin_nontemporal_load(&X[gi]) * areg[r];
            __builtin_nontemporal_store(xv, &Xtilde[gi]);
            Ax[r * AXS + tid] = f2bf(xv);
        }
        __syncthreads();

        // ---- accumulators start at bias (b_ih + b_hh)
        v8f acc[4][2];
        #pragma unroll
        for (int g = 0; g < 4; ++g)
            #pragma unroll
            for (int h2 = 0; h2 < 2; ++h2)
                #pragma unroll
                for (int r = 0; r < 8; ++r) acc[g][h2][r] = biasv[g][h2];

        // ---- x~ @ W_ih^T  (K = N_)
        for (int k0 = 0; k0 < N_; k0 += 32) {
            int kk = k0;
            NOHOIST_I32(kk);   // opaque K offset: loads stay in-loop, stay global
            v16bf a;
            *(v8bf*)&a       = *(const v8bf*)&Ax[rowA * AXS + k0 + kbA];
            *((v8bf*)&a + 1) = *(const v8bf*)&Ax[rowA * AXS + k0 + kbA + 8];
            #pragma unroll
            for (int g = 0; g < 4; ++g)
                #pragma unroll
                for (int h2 = 0; h2 < 2; ++h2) {
                    v16bf bf = *(const v16bf*)(WihBf + colOff[g][h2] + kk);
                    acc[g][h2] = __builtin_amdgcn_wmma_f32_16x16x32_bf16(
                        false, a, false, bf, (short)0, acc[g][h2], false, false);
                }
        }
        // ---- h @ W_hh^T  (K = H_)
        for (int k0 = 0; k0 < H_; k0 += 32) {
            int kk = k0;
            NOHOIST_I32(kk);
            v16bf a;
            *(v8bf*)&a       = *(const v8bf*)&Ah[rowA * AXS + k0 + kbA];
            *((v8bf*)&a + 1) = *(const v8bf*)&Ah[rowA * AXS + k0 + kbA + 8];
            #pragma unroll
            for (int g = 0; g < 4; ++g)
                #pragma unroll
                for (int h2 = 0; h2 < 2; ++h2) {
                    v16bf bf = *(const v16bf*)(WhhBf + colOff[g][h2] + kk);
                    acc[g][h2] = __builtin_amdgcn_wmma_f32_16x16x32_bf16(
                        false, a, false, bf, (short)0, acc[g][h2], false, false);
                }
        }

        __syncthreads();   // all waves finished reading old Ah

        // ---- LSTM pointwise in accumulator layout:
        // element r, lane L -> row m = r + 8*(L>=16), hidden = 32w+16*h2+(L&15)
        #pragma unroll
        for (int h2 = 0; h2 < 2; ++h2) {
            int hh = wv * 32 + h2 * 16 + ln16;
            #pragma unroll
            for (int r = 0; r < 8; ++r) {
                int m = r + hi * 8;
                float iv = sigmoidf(acc[0][h2][r]);
                float fv = sigmoidf(acc[1][h2][r]);
                float gv = tanhf(acc[2][h2][r]);
                float ov = sigmoidf(acc[3][h2][r]);
                float c  = fv * cfrag[h2][r] + iv * gv;
                cfrag[h2][r] = c;
                float hval = ov * tanhf(c);
                __builtin_nontemporal_store(
                    hval, &Xenc[(size_t)(b0 + m) * TM1 * H_ + (size_t)t * H_ + hh]);
                Ah[m * AXS + hh] = f2bf(hval);
            }
        }
        __syncthreads();   // new Ah visible; also guards Ax rewrite next step
    }
}

// ---------------------------------------------------------------------------
extern "C" void kernel_launch(void* const* d_in, const int* in_sizes, int n_in,
                              void* d_out, int out_size, void* d_ws, size_t ws_size,
                              hipStream_t stream) {
    const float* X      = (const float*)d_in[0];
    const float* attn_w = (const float*)d_in[1];
    const float* attn_b = (const float*)d_in[2];
    const float* W_ih   = (const float*)d_in[3];
    const float* W_hh   = (const float*)d_in[4];
    const float* b_ih   = (const float*)d_in[5];
    const float* b_hh   = (const float*)d_in[6];

    float* Xtilde = (float*)d_out;
    float* Xenc   = Xtilde + (size_t)B_ * TM1 * N_;

    // workspace: alpha (256KB) | W_ih bf16 (512KB) | W_hh bf16 (512KB)
    char* ws = (char*)d_ws;
    float*          alphaW = (float*)ws;
    unsigned short* WihBf  = (unsigned short*)(ws + 262144);
    unsigned short* WhhBf  = (unsigned short*)(ws + 262144 + 524288);

    alpha_kernel<<<B_, 256, 0, stream>>>(X, attn_w, attn_b, alphaW);
    cvt_kernel<<<(4 * H_ * N_) / 256, 256, 0, stream>>>(W_ih, W_hh, WihBf, WhhBf);
    lstm_kernel<<<B_ / 16, 256, 0, stream>>>(X, alphaW, WihBf, WhhBf,
                                             b_ih, b_hh, Xtilde, Xenc);
}
